// YOLO_2130303778844
// MI455X (gfx1250) — compile-verified, matrix operand order checked by reading, not run on hardware
//
#include <hip/hip_runtime.h>
#include <hip/hip_bf16.h>
#include <math.h>

typedef __attribute__((ext_vector_type(16))) _Float16 v16h;
typedef __attribute__((ext_vector_type(8)))  float    v8f;
typedef int v4i_ __attribute__((vector_size(16)));   // matches async-LDS builtin param

#define NDET   1805
#define NCLS   80
#define NHW    361
#define NANCH  5
#define GS     19
#define STRIDEF 32.0f
#define CONF_TH 0.01f
#define NMS_TH  0.5f
#define SORT_N 2048

__device__ __forceinline__ float sigmoidf_(float x) {
    return 1.0f / (1.0f + __expf(-x));
}

// ---------------------------------------------------------------------------
// Kernel 1: fused box decode + objectness sigmoid + softmax(max/argmax/sum).
// Each wave32 owns 16 detection rows; the softmax denominator (row-sums of
// exp(x - rowmax) over 80 classes) is computed as (16x96)·ones via three
// V_WMMA_F32_16X16X32_F16 ops, with the A-matrix packed per the CDNA5 ISA
// 16-bit A layout. scores = sigmoid(conf) / rowsum, cls = argmax.
// ---------------------------------------------------------------------------
__global__ void yolo_decode_score_kernel(const float* __restrict__ conf,
                                         const float* __restrict__ cls,
                                         const float* __restrict__ txty,
                                         const float* __restrict__ anchors,
                                         float* __restrict__ boxes_out,
                                         float* __restrict__ scores_out,
                                         int*   __restrict__ cls_out) {
    const int lane  = threadIdx.x & 31;
    const int half  = lane >> 4;               // 0: K-low half, 1: K-high half
    const int wave  = blockIdx.x * (blockDim.x >> 5) + (threadIdx.x >> 5);
    const int row   = wave * 16 + (lane & 15); // logical detection index
    const int rc    = row < NDET ? row : (NDET - 1);  // clamped for loads

    const float* crow = cls + rc * NCLS;

    // ---- max / argmax over this row (lane pair splits 80 classes 40/40) ----
    float m = -INFINITY; int mi = 0;
    for (int t = 0; t < 40; ++t) {
        int c = half * 40 + t;
        float v = crow[c];
        if (v > m) { m = v; mi = c; }
    }
    float om  = __shfl_xor(m, 16);
    int   omi = __shfl_xor(mi, 16);
    float rowmax; int rowarg;
    if (om > m || (om == m && omi < mi)) { rowmax = om; rowarg = omi; }
    else                                 { rowmax = m;  rowarg = mi;  }

    // ---- softmax denominator via WMMA: rowsum = sum_k exp(x_k - rowmax) ----
    v16h ones;
    #pragma unroll
    for (int t = 0; t < 16; ++t) ones[t] = (_Float16)1.0f;

    v8f acc = {};
    #pragma unroll
    for (int kb = 0; kb < 96; kb += 32) {
        v16h a;
        #pragma unroll
        for (int t = 0; t < 16; ++t) {
            // CDNA5 16-bit A-matrix 16x32 layout:
            // lanes 0-15:  K = {0..7, 16..23};  lanes 16-31: K = {8..15, 24..31}
            int klocal = (t < 8 ? t : t + 8) + half * 8;
            int k = kb + klocal;
            float e = (k < NCLS) ? __expf(crow[k] - rowmax) : 0.0f;
            a[t] = (_Float16)e;
        }
        acc = __builtin_amdgcn_wmma_f32_16x16x32_f16(
            /*neg_a=*/false, a, /*neg_b=*/false, ones,
            /*c_mod=*/(short)0, acc, /*reuse_a=*/false, /*reuse_b=*/false);
    }

    // D columns are identical; lanes 0-15 hold rows 0-7 in v0..7, lanes 16-31
    // hold rows 8-15. Exchange halves, then select this lane's row sum.
    float sums[16];
    #pragma unroll
    for (int i = 0; i < 8; ++i) {
        float mine   = acc[i];
        float theirs = __shfl_xor(mine, 16);
        sums[i]     = half ? theirs : mine;
        sums[i + 8] = half ? mine   : theirs;
    }
    const int rr = lane & 15;
    float rowsum = sums[0];
    #pragma unroll
    for (int i = 1; i < 16; ++i) rowsum = (rr == i) ? sums[i] : rowsum;

    float obj   = sigmoidf_(conf[rc]);
    float score = obj / rowsum;   // softmax max term is exp(0) = 1

    // ---- box decode ----
    const int a_i = rc % NANCH;
    const int hw  = rc / NANCH;
    const float gx = (float)(hw % GS);
    const float gy = (float)(hw / GS);
    const float* t4 = txty + (hw * NANCH + a_i) * 4;
    float cx = (sigmoidf_(t4[0]) + gx) * STRIDEF;
    float cy = (sigmoidf_(t4[1]) + gy) * STRIDEF;
    float w  = __expf(t4[2]) * anchors[a_i * 2 + 0] * STRIDEF;
    float h  = __expf(t4[3]) * anchors[a_i * 2 + 1] * STRIDEF;
    const float inv = 1.0f / (GS * STRIDEF);   // 1/608
    float x1 = fminf(fmaxf((cx - 0.5f * w) * inv, 0.0f), 1.0f);
    float y1 = fminf(fmaxf((cy - 0.5f * h) * inv, 0.0f), 1.0f);
    float x2 = fminf(fmaxf((cx + 0.5f * w) * inv, 0.0f), 1.0f);
    float y2 = fminf(fmaxf((cy + 0.5f * h) * inv, 0.0f), 1.0f);

    if (row < NDET && half == 0) {
        boxes_out[row * 4 + 0] = x1;
        boxes_out[row * 4 + 1] = y1;
        boxes_out[row * 4 + 2] = x2;
        boxes_out[row * 4 + 3] = y2;
        scores_out[row] = score;
        cls_out[row]    = rowarg;
    }
}

// ---------------------------------------------------------------------------
// Kernel 2: single-workgroup greedy per-class NMS, all state in LDS.
// Box data is staged into LDS with gfx1250 async global->LDS B128 transfers
// (ASYNCcnt / s_wait_asynccnt) when available, overlapping with the rest of
// the LDS init. Then: bitonic sort (desc score, index tie-break) + sequential
// greedy loop with the inner suppression scan parallelized over 1024 threads.
// ---------------------------------------------------------------------------
__global__ void __launch_bounds__(1024)
yolo_nms_kernel(const float* __restrict__ boxes,
                const float* __restrict__ scores,
                const int*   __restrict__ cls_ind,
                int* __restrict__ keep_out) {
    __shared__ float  skey[SORT_N];
    __shared__ int    sidx[SORT_N];
    __shared__ float4 sbox[NDET];      // interleaved x1,y1,x2,y2
    __shared__ int    scls[NDET];
    __shared__ int    skeep[NDET];

    const int tid = threadIdx.x;
    const float4* gbox = reinterpret_cast<const float4*>(boxes);

    for (int i = tid; i < NDET; i += blockDim.x) {
#if defined(__AMDGCN__) && __has_builtin(__builtin_amdgcn_global_load_async_to_lds_b128)
        // gfx1250 async DMA: memory -> LDS, no VGPR round-trip (ASYNCcnt).
        __builtin_amdgcn_global_load_async_to_lds_b128(
            (__attribute__((address_space(1))) v4i_*)(gbox + i),
            (__attribute__((address_space(3))) v4i_*)(&sbox[i]),
            /*imm offset=*/0, /*cpol=*/0);
#else
        sbox[i] = gbox[i];
#endif
        float s = scores[i];
        skey[i]  = s;
        sidx[i]  = i;
        scls[i]  = cls_ind[i];
        skeep[i] = (s >= CONF_TH) ? 1 : 0;
    }
    for (int i = NDET + tid; i < SORT_N; i += blockDim.x) {
        skey[i] = -INFINITY; sidx[i] = i;
    }
#if defined(__AMDGCN__)
#if __has_builtin(__builtin_amdgcn_s_wait_asynccnt)
    __builtin_amdgcn_s_wait_asynccnt(0);
#else
    asm volatile("s_wait_asynccnt 0" ::: "memory");
#endif
#endif
    __syncthreads();

    // Bitonic sort: descending by score, ascending index on ties (stable-ish).
    for (int k = 2; k <= SORT_N; k <<= 1) {
        for (int j = k >> 1; j > 0; j >>= 1) {
            for (int t = tid; t < SORT_N; t += blockDim.x) {
                int ixj = t ^ j;
                if (ixj > t) {
                    bool up = ((t & k) == 0);
                    float ka = skey[t], kb = skey[ixj];
                    int   ia = sidx[t], ib = sidx[ixj];
                    bool before = (ka > kb) || (ka == kb && ia < ib);
                    if (up ? !before : before) {
                        skey[t] = kb; skey[ixj] = ka;
                        sidx[t] = ib; sidx[ixj] = ia;
                    }
                }
            }
            __syncthreads();
        }
    }

    // Greedy suppression in sorted order (per-class via cls match).
    for (int i = 0; i < NDET; ++i) {
        int bi = sidx[i];
        if (skeep[bi]) {                 // uniform across block
            float4 B = sbox[bi];
            float areaI = (B.z - B.x) * (B.w - B.y);
            int   ci    = scls[bi];
            for (int t = i + 1 + tid; t < NDET; t += blockDim.x) {
                int bj = sidx[t];
                if (skeep[bj] && scls[bj] == ci) {
                    float4 J = sbox[bj];
                    float xx1 = fmaxf(B.x, J.x);
                    float yy1 = fmaxf(B.y, J.y);
                    float xx2 = fminf(B.z, J.z);
                    float yy2 = fminf(B.w, J.w);
                    // exact reference semantics: 1e-28 floor, no zero clamp
                    float inter = fmaxf(1e-28f, xx2 - xx1) * fmaxf(1e-28f, yy2 - yy1);
                    float areaJ = (J.z - J.x) * (J.w - J.y);
                    float iou   = inter / (areaI + areaJ - inter);
                    if (iou > NMS_TH) skeep[bj] = 0;
                }
            }
        }
        __syncthreads();
    }

    for (int i = tid; i < NDET; i += blockDim.x) keep_out[i] = skeep[i];
}

// ---------------------------------------------------------------------------
// Kernel 3: assemble outputs: [boxes|score*keep] (Nx5 f32), cls_inds (int32),
// keep (as 0/1).
// ---------------------------------------------------------------------------
__global__ void yolo_output_kernel(const float* __restrict__ boxes,
                                   const float* __restrict__ scores,
                                   const int*   __restrict__ cls_ind,
                                   const int*   __restrict__ keep,
                                   float* __restrict__ out) {
    int n = blockIdx.x * blockDim.x + threadIdx.x;
    if (n >= NDET) return;
    out[n * 5 + 0] = boxes[n * 4 + 0];
    out[n * 5 + 1] = boxes[n * 4 + 1];
    out[n * 5 + 2] = boxes[n * 4 + 2];
    out[n * 5 + 3] = boxes[n * 4 + 3];
    out[n * 5 + 4] = scores[n] * (keep[n] ? 1.0f : 0.0f);
    int* ci_out = reinterpret_cast<int*>(out + NDET * 5);
    ci_out[n] = cls_ind[n];
    float* keep_f = out + NDET * 5 + NDET;
    keep_f[n] = keep[n] ? 1.0f : 0.0f;
}

extern "C" void kernel_launch(void* const* d_in, const int* in_sizes, int n_in,
                              void* d_out, int out_size, void* d_ws, size_t ws_size,
                              hipStream_t stream) {
    const float* conf    = (const float*)d_in[0];   // (1, 1805, 1)
    const float* cls     = (const float*)d_in[1];   // (1, 1805, 80)
    const float* txty    = (const float*)d_in[2];   // (1, 361, 5, 4)
    const float* anchors = (const float*)d_in[3];   // (5, 2)

    float* wsf      = (float*)d_ws;
    float* ws_boxes = wsf;                  // NDET*4
    float* ws_score = wsf + NDET * 4;       // NDET
    int*   ws_cls   = (int*)(wsf + NDET * 5);
    int*   ws_keep  = (int*)(wsf + NDET * 6);

    // Kernel 1: 16 rows per wave, 8 waves (128 rows) per 256-thread block.
    int nwaves  = (NDET + 15) / 16;
    int nblocks = (nwaves + 7) / 8;
    yolo_decode_score_kernel<<<nblocks, 256, 0, stream>>>(
        conf, cls, txty, anchors, ws_boxes, ws_score, ws_cls);

    // Kernel 2: single-workgroup LDS-resident NMS.
    yolo_nms_kernel<<<1, 1024, 0, stream>>>(ws_boxes, ws_score, ws_cls, ws_keep);

    // Kernel 3: output assembly.
    yolo_output_kernel<<<(NDET + 255) / 256, 256, 0, stream>>>(
        ws_boxes, ws_score, ws_cls, ws_keep, (float*)d_out);
}
